// PerformerLinearAttention_70995809402910
// MI455X (gfx1250) — compile-verified, matrix operand order checked by reading, not run on hardware
//
#include <hip/hip_runtime.h>
#include <hip/hip_bf16.h>
#include <stdint.h>

// ---------------- problem constants (compile-time) ----------------
#define B_    4
#define S_    4096
#define HID_  2048
#define HD_   64
#define NH_   32
#define NKV_  8
#define NF_   128
#define QKV_  ((NH_ + 2 * NKV_) * HD_)   // 3072
#define EPS_  1e-4f

// GEMM tiling: 128x128 block tile, K-step 32 (bf16 wmma K), 256 thr = 8 waves
#define BM 128
#define BN 128
#define BK 32
#define LDT 40   // ushort pitch per LDS row -> 80B: 16B aligned, stride-20 banks (conflict free)

typedef __attribute__((ext_vector_type(16))) __bf16 v16bf;
typedef __attribute__((ext_vector_type(8)))  float  v8f;
typedef __attribute__((ext_vector_type(8)))  unsigned int v8u;

union FragU { v8u u; v16bf b; };

__device__ __forceinline__ unsigned short f2bf(float x) {
  unsigned int u = __builtin_bit_cast(unsigned int, x);
  u += 0x7FFFu + ((u >> 16) & 1u);           // round-to-nearest-even
  return (unsigned short)(u >> 16);
}
__device__ __forceinline__ float bf2f(unsigned short h) {
  return __builtin_bit_cast(float, (unsigned int)h << 16);
}
__device__ __forceinline__ unsigned pack2bf(float a, float b) {
  return (unsigned)f2bf(a) | ((unsigned)f2bf(b) << 16);
}

// ---- CDNA5 async global->LDS copy (ASYNCcnt-tracked), 32B per lane ----
__device__ __forceinline__ void async_copy32(const unsigned short* lds_dst,
                                             const unsigned short* gsrc) {
  unsigned lds_off = (unsigned)reinterpret_cast<unsigned long long>(lds_dst);
  unsigned long long ga = reinterpret_cast<unsigned long long>(gsrc);
  asm volatile(
      "global_load_async_to_lds_b128 %0, %1, off\n\t"
      "global_load_async_to_lds_b128 %0, %1, off offset:16"
      :: "v"(lds_off), "v"(ga) : "memory");
}
__device__ __forceinline__ void wait_async0() {
  asm volatile("s_wait_asynccnt 0" ::: "memory");
}

// Gather one 16x32 bf16 operand row-fragment from LDS following the CDNA5
// 16-bit A/B VGPR layout: lanes 0-15 hold K pairs {0..7,16..23},
// lanes 16-31 hold {8..15,24..31}; each VGPR = 2 consecutive K (one dword).
__device__ __forceinline__ v16bf frag_ld(const unsigned short* rowp, int hi) {
  const unsigned int* u = (const unsigned int*)rowp;   // row base 16B aligned
  FragU f;
  const int base = hi ? 4 : 0;
#pragma unroll
  for (int i = 0; i < 4; ++i) {
    f.u[i]     = u[base + i];       // K pairs 0..7   (lo) / 8..15  (hi)
    f.u[4 + i] = u[8 + base + i];   // K pairs 16..23 (lo) / 24..31 (hi)
  }
  return f.b;
}

struct RopeArgs {
  const unsigned short* qkv;  // (B*S, QKV_) bf16
  const float* cosb;          // (S, HD)
  const float* sinb;          // (S, HD)
  int nheads;                 // NH_ (q) or NKV_ (k)
  int head_off;               // 0 (q) or NH_*HD_ (k)
};

// bf16-in / bf16-wmma / {f32|bf16}-out tile GEMM:  C = epi(A @ B^T)
//  A: (M,K) bf16 row-major (or RoPE-synthesized), B: (N,K) bf16 row-major.
//  Double-buffered LDS, async global->LDS staging, one barrier per K-step.
__device__ __forceinline__ void gemm_core(
    const unsigned short* __restrict__ A, int lda, int rope, RopeArgs ra,
    const unsigned short* __restrict__ Bm, int ldb, int Nvalid,
    void* __restrict__ Cout, int ldc, int out_bf16, int K,
    const float* __restrict__ den, int den_stride,
    float escale, float eadd, int do_relu,
    int bm, int bn) {
  __shared__ alignas(16) unsigned short sA[2][BM * LDT];
  __shared__ alignas(16) unsigned short sB[2][BN * LDT];

  const int t    = threadIdx.x;
  const int lane = t & 31;
  const int wave = t >> 5;
  const int wm   = wave & 3;   // 4 wave rows  -> 32 M each
  const int wn   = wave >> 2;  // 2 wave cols  -> 64 N each
  const int r0   = bm * BM;
  const int c0   = bn * BN;
  const int row  = t >> 1;          // cooperative loader: half-row per thread
  const int ch   = (t & 1) * 16;

  v8f acc[2][4];
#pragma unroll
  for (int m = 0; m < 2; ++m)
#pragma unroll
    for (int n = 0; n < 4; ++n)
#pragma unroll
      for (int i = 0; i < 8; ++i) acc[m][n][i] = 0.0f;

  auto loadA = [&](int buf, int k0) {
    unsigned short* dst = &sA[buf][row * LDT + ch];
    if (!rope) {
      async_copy32(dst, A + (size_t)(r0 + row) * lda + k0 + ch);
    } else {
      // A row r = (b*S+s)*nheads + h ; element d in [0,64): RoPE on the fly
      const int r  = r0 + row;
      const int bs = r / ra.nheads;
      const int h  = r - bs * ra.nheads;
      const int s  = bs & (S_ - 1);
      const unsigned short* x = ra.qkv + (size_t)bs * QKV_ + ra.head_off + h * HD_;
      const float* cp = ra.cosb + (size_t)s * HD_;
      const float* sp = ra.sinb + (size_t)s * HD_;
      unsigned int* d4 = (unsigned int*)dst;
#pragma unroll
      for (int j = 0; j < 16; j += 2) {
        const int d0 = k0 + ch + j;
        const int d1 = d0 + 1;
        const float ra0 = (d0 < HD_ / 2) ? -bf2f(x[d0 + HD_ / 2]) : bf2f(x[d0 - HD_ / 2]);
        const float ra1 = (d1 < HD_ / 2) ? -bf2f(x[d1 + HD_ / 2]) : bf2f(x[d1 - HD_ / 2]);
        d4[j >> 1] = pack2bf(bf2f(x[d0]) * cp[d0] + ra0 * sp[d0],
                             bf2f(x[d1]) * cp[d1] + ra1 * sp[d1]);
      }
    }
  };
  auto loadB = [&](int buf, int k0) {
    unsigned short* dst = &sB[buf][row * LDT + ch];
    const int n = c0 + row;
    if (n < Nvalid) {
      async_copy32(dst, Bm + (size_t)n * ldb + k0 + ch);
    } else {
      uint4* d4 = (uint4*)dst;
      d4[0] = make_uint4(0, 0, 0, 0);
      d4[1] = make_uint4(0, 0, 0, 0);
    }
  };

  int cur = 0;
  loadA(0, 0);
  loadB(0, 0);
  const int lm = lane & 15;
  const int hi = lane >> 4;

  for (int k0 = 0; k0 < K; k0 += BK) {
    wait_async0();       // own async copies into buf[cur] done
    __syncthreads();     // everyone's copies visible; buf[cur^1] provably dead
    const int nxt = k0 + BK;
    if (nxt < K) { loadA(cur ^ 1, nxt); loadB(cur ^ 1, nxt); }

    v16bf af[2], bfr[4];
#pragma unroll
    for (int mt = 0; mt < 2; ++mt)
      af[mt] = frag_ld(&sA[cur][(wm * 32 + mt * 16 + lm) * LDT], hi);
#pragma unroll
    for (int nt = 0; nt < 4; ++nt)
      bfr[nt] = frag_ld(&sB[cur][(wn * 64 + nt * 16 + lm) * LDT], hi);
#pragma unroll
    for (int mt = 0; mt < 2; ++mt)
#pragma unroll
      for (int nt = 0; nt < 4; ++nt)
        acc[mt][nt] = __builtin_amdgcn_wmma_f32_16x16x32_bf16(
            false, af[mt], false, bfr[nt], (short)0, acc[mt][nt], false, false);
    cur ^= 1;
  }

  // ---- epilogue: C/D layout: VGPR r -> (M = r + hi*8, N = lane&15) ----
#pragma unroll
  for (int mt = 0; mt < 2; ++mt)
#pragma unroll
    for (int nt = 0; nt < 4; ++nt) {
      const int col = c0 + wn * 64 + nt * 16 + lm;
      if (col < Nvalid) {
#pragma unroll
        for (int rI = 0; rI < 8; ++rI) {
          const int r = r0 + wm * 32 + mt * 16 + hi * 8 + rI;
          float v = acc[mt][nt][rI] * escale;
          if (do_relu) v = fmaxf(v, 0.0f);
          v += eadd;
          if (den) v /= den[(size_t)r * den_stride];
          if (out_bf16)
            ((unsigned short*)Cout)[(size_t)r * ldc + col] = f2bf(v);
          else
            ((float*)Cout)[(size_t)r * ldc + col] = v;
        }
      }
    }
}

// ---------------- kernels ----------------
__global__ void k_cvt_bf16(const float* __restrict__ src,
                           unsigned short* __restrict__ dst, long n4) {
  const long i = (long)blockIdx.x * blockDim.x + threadIdx.x;  // float4 units
  if (i < n4) {
    float4 v = ((const float4*)src)[i];
    uint2 p;
    p.x = pack2bf(v.x, v.y);
    p.y = pack2bf(v.z, v.w);
    ((uint2*)dst)[i] = p;
  }
}

__global__ void k_gemm_plain(const unsigned short* A, int lda,
                             const unsigned short* Bm, int ldb, int Nvalid,
                             void* C, int ldc, int out_bf16, int K) {
  RopeArgs ra{};
  gemm_core(A, lda, 0, ra, Bm, ldb, Nvalid, C, ldc, out_bf16, K,
            nullptr, 0, 1.0f, 0.0f, 0, blockIdx.x, blockIdx.y);
}

__global__ void k_gemm_phi(const unsigned short* qkv, const float* cosb,
                           const float* sinb, int nheads, int head_off,
                           const unsigned short* proj, unsigned short* C,
                           float eadd) {
  RopeArgs ra{qkv, cosb, sinb, nheads, head_off};
  const float ratio = 0.08838834764831845f;        // 1/sqrt(128)
  const float norm  = 1.0f / (8.0f + EPS_);        // 1/(sqrt(64)+eps)
  gemm_core(nullptr, 0, 1, ra, proj, HD_, NF_, C, NF_, 1, HD_,
            nullptr, 0, ratio * norm, eadd, 1, blockIdx.x, blockIdx.y);
}

// num/den: for head (b,n): C(S,64) = qp(S,128) @ kvT[b,g]^T, divided by den
__global__ void k_gemm_num(const unsigned short* qp, const unsigned short* kvT,
                           const float* den, unsigned short* attn) {
  const int z = blockIdx.z;
  const int b = z / NH_;
  const int n = z % NH_;
  const int g = n / (NH_ / NKV_);
  const unsigned short* A  = qp  + ((size_t)((size_t)b * S_) * NH_ + n) * NF_;
  const unsigned short* Bm = kvT + (size_t)(b * NKV_ + g) * HD_ * NF_;
  unsigned short* C        = attn + (size_t)b * S_ * HID_ + n * HD_;
  const float* d           = den + (size_t)b * S_ * NH_ + n;
  RopeArgs ra{};
  gemm_core(A, NH_ * NF_, 0, ra, Bm, NF_, HD_, C, HID_, 1, NF_,
            d, NH_, 1.0f, 0.0f, 0, blockIdx.x, blockIdx.y);
}

// kv reduction: one block per (b,g): kvT[b,g](64x128) = sum_s v(s)^T kp(s);
// z[b,g](128) = sum_s kp.  M=f=128, N=h=64, K=s=4096.  All-bf16 staging.
__global__ void k_kv_z(const unsigned short* __restrict__ kp,
                       const unsigned short* __restrict__ qkv,
                       unsigned short* __restrict__ kvT,
                       float* __restrict__ zv) {
  __shared__ alignas(16) unsigned short sA[NF_ * LDT];  // 128 f x 32 s
  __shared__ alignas(16) unsigned short sB[HD_ * LDT];  // 64  h x 32 s
  const int z = blockIdx.x;
  const int b = z / NKV_;
  const int g = z % NKV_;
  const int t    = threadIdx.x;
  const int lane = t & 31;
  const int wave = t >> 5;     // wave -> 16-row (f) slab; all 4 n-tiles
  const int voff = (NH_ + NKV_) * HD_ + g * HD_;

  v8f acc[4];
#pragma unroll
  for (int n = 0; n < 4; ++n)
#pragma unroll
    for (int i = 0; i < 8; ++i) acc[n][i] = 0.0f;
  float zacc = 0.0f;

  for (int s0 = 0; s0 < S_; s0 += BK) {
    {  // A tile: kp (bf16), transpose-on-store (f-major LDS rows, s along K)
      const int sc = t >> 3;            // 0..31 (s)
      const int f0 = (t & 7) * 16;      // 16 contiguous f
      const unsigned short* src =
          kp + ((size_t)((size_t)b * S_ + s0 + sc) * NKV_ + g) * NF_ + f0;
#pragma unroll
      for (int j = 0; j < 16; ++j) sA[(f0 + j) * LDT + sc] = src[j];
    }
    {  // B tile: v from qkv buffer (bf16)
      const int sc = t >> 3;
      const int h0 = (t & 7) * 8;
      const unsigned short* src =
          qkv + (size_t)((size_t)b * S_ + s0 + sc) * QKV_ + voff + h0;
#pragma unroll
      for (int j = 0; j < 8; ++j) sB[(h0 + j) * LDT + sc] = src[j];
    }
    __syncthreads();

    if (t < NF_) {  // z accumulation (re-converges before wmma)
#pragma unroll
      for (int s = 0; s < BK; ++s) zacc += bf2f(sA[t * LDT + s]);
    }
    const int lm = lane & 15;
    const int hi = lane >> 4;
    v16bf af = frag_ld(&sA[(wave * 16 + lm) * LDT], hi);
#pragma unroll
    for (int nt = 0; nt < 4; ++nt) {
      v16bf bfr = frag_ld(&sB[(nt * 16 + lm) * LDT], hi);
      acc[nt] = __builtin_amdgcn_wmma_f32_16x16x32_bf16(
          false, af, false, bfr, (short)0, acc[nt], false, false);
    }
    __syncthreads();
  }

  // store transposed: kvT[b,g][h][f] as bf16, packed 8 consecutive f per lane
  unsigned short* out = kvT + (size_t)(b * NKV_ + g) * HD_ * NF_;
  const int lm = lane & 15;
  const int hi = lane >> 4;
#pragma unroll
  for (int nt = 0; nt < 4; ++nt) {
    uint4 p;
    p.x = pack2bf(acc[nt][0], acc[nt][1]);
    p.y = pack2bf(acc[nt][2], acc[nt][3]);
    p.z = pack2bf(acc[nt][4], acc[nt][5]);
    p.w = pack2bf(acc[nt][6], acc[nt][7]);
    *(uint4*)&out[(size_t)(nt * 16 + lm) * NF_ + (wave * 16 + hi * 8)] = p;
  }
  if (t < NF_) zv[(size_t)(b * NKV_ + g) * NF_ + t] = zacc;
}

__global__ void k_den(const unsigned short* __restrict__ qp,
                      const float* __restrict__ zv, float* __restrict__ den) {
  const int r = blockIdx.x * blockDim.x + threadIdx.x;  // (b*S+s)*NH + n
  if (r >= B_ * S_ * NH_) return;
  const int n  = r % NH_;
  const int bs = r / NH_;
  const int b  = bs / S_;
  const int g  = n / (NH_ / NKV_);
  const unsigned int* q = (const unsigned int*)(qp + (size_t)r * NF_);
  const float* zp = zv + (size_t)(b * NKV_ + g) * NF_;
  float acc = 0.0f;
#pragma unroll
  for (int i = 0; i < NF_ / 2; ++i) {
    const unsigned u = q[i];
    acc += bf2f((unsigned short)u) * zp[2 * i] +
           bf2f((unsigned short)(u >> 16)) * zp[2 * i + 1];
  }
  den[r] = acc + 1e-6f;
}

// ---------------- host launch ----------------
extern "C" void kernel_launch(void* const* d_in, const int* in_sizes, int n_in,
                              void* d_out, int out_size, void* d_ws, size_t ws_size,
                              hipStream_t stream) {
  const float* cosb   = (const float*)d_in[0];
  const float* sinb   = (const float*)d_in[1];
  const float* hidden = (const float*)d_in[2];
  const float* w_qkv  = (const float*)d_in[3];
  const float* w_o    = (const float*)d_in[4];
  const float* proj   = (const float*)d_in[5];

  const size_t nHid  = (size_t)B_ * S_ * HID_;          // 33554432
  const size_t nWqkv = (size_t)QKV_ * HID_;             // 6291456
  const size_t nWo   = (size_t)HID_ * NH_ * HD_;        // 4194304
  const size_t nProj = (size_t)NF_ * HD_;               // 8192
  const size_t nQkv  = (size_t)B_ * S_ * QKV_;          // 50331648
  const size_t nQp   = (size_t)B_ * S_ * NH_ * NF_;     // 67108864
  const size_t nKp   = (size_t)B_ * S_ * NKV_ * NF_;    // 16777216
  const size_t nKvT  = (size_t)B_ * NKV_ * HD_ * NF_;   // 262144

  unsigned short* hid_bf  = (unsigned short*)d_ws;
  unsigned short* wqkv_bf = hid_bf + nHid;
  unsigned short* wo_bf   = wqkv_bf + nWqkv;
  unsigned short* proj_bf = wo_bf + nWo;
  unsigned short* qkv_bf  = proj_bf + nProj;
  unsigned short* qp_bf   = qkv_bf + nQkv;
  unsigned short* kp_bf   = qp_bf + nQp;
  unsigned short* kvT_bf  = kp_bf + nKp;
  float* zv  = (float*)(kvT_bf + nKvT);
  float* den = zv + (size_t)B_ * NKV_ * NF_;
  unsigned short* attn_bf = qkv_bf;  // reuse: qkv dead after k_kv_z

  const dim3 blk(256);
  const int Mrows = B_ * S_;  // 16384

  // 0) one-time f32 -> bf16 operand staging (bandwidth-bound, converts once)
  k_cvt_bf16<<<dim3(nHid / 4 / 256), blk, 0, stream>>>(hidden, hid_bf, nHid / 4);
  k_cvt_bf16<<<dim3(nWqkv / 4 / 256), blk, 0, stream>>>(w_qkv, wqkv_bf, nWqkv / 4);
  k_cvt_bf16<<<dim3(nWo / 4 / 256), blk, 0, stream>>>(w_o, wo_bf, nWo / 4);
  k_cvt_bf16<<<dim3(nProj / 4 / 256 + 1), blk, 0, stream>>>(proj, proj_bf, nProj / 4);

  // 1) qkv = hidden @ w_qkv.T   (bf16 out)
  k_gemm_plain<<<dim3(Mrows / BM, QKV_ / BN), blk, 0, stream>>>(
      hid_bf, HID_, wqkv_bf, HID_, QKV_, qkv_bf, QKV_, 1, HID_);
  // 2) qp = relu(scale * rope(q) @ proj.T) + eps
  k_gemm_phi<<<dim3((Mrows * NH_) / BM, 1), blk, 0, stream>>>(
      qkv_bf, cosb, sinb, NH_, 0, proj_bf, qp_bf, EPS_);
  // 3) kp per KV head (GQA)
  k_gemm_phi<<<dim3((Mrows * NKV_) / BM, 1), blk, 0, stream>>>(
      qkv_bf, cosb, sinb, NKV_, NH_ * HD_, proj_bf, kp_bf, 0.0f);
  // 4) kvT, z
  k_kv_z<<<dim3(B_ * NKV_), blk, 0, stream>>>(kp_bf, qkv_bf, kvT_bf, zv);
  // 5) den
  k_den<<<dim3((Mrows * NH_) / 256), blk, 0, stream>>>(qp_bf, zv, den);
  // 6) attn = (qp @ kv) / den, per (b, head)  (bf16 out)
  k_gemm_num<<<dim3(S_ / BM, 1, B_ * NH_), blk, 0, stream>>>(qp_bf, kvT_bf, den, attn_bf);
  // 7) out = attn @ w_o.T  (f32 out)
  k_gemm_plain<<<dim3(Mrows / BM, HID_ / BN), blk, 0, stream>>>(
      attn_bf, HID_, wo_bf, HID_, HID_, d_out, HID_, 0, HID_);

  (void)in_sizes; (void)n_in; (void)out_size; (void)ws_size;
}
